// TargetModel_9122510536839
// MI455X (gfx1250) — matrix operand order repaired and checked by reading, not compiled, not run on hardware
//
#include <hip/hip_runtime.h>
#include <hip/hip_bf16.h>

// ---------------------------------------------------------------------------
// GNN layer for MI455X (gfx1250, wave32, WMMA):
//   edge MLP (2x 256x256 GEMM, bf16 WMMA, fp32 acc) -> atomic scatter-add
//   node MLP (512x512 + 512x128 GEMM, bf16 WMMA)   -> batchnorm
// Edge kernel uses 64-edge blocks (8 waves, 2x4 wave grid, 32x64 slab/wave)
// so each B fragment is reused across 2 M-tiles -> halves L2 weight traffic.
// ---------------------------------------------------------------------------

typedef __bf16 bf16_t;
typedef __attribute__((ext_vector_type(16))) __bf16 v16bf;
typedef __attribute__((ext_vector_type(8)))  __bf16 v8bf;
typedef __attribute__((ext_vector_type(8)))  float  v8f;

#define LDIM   128          // lifted dim
#define MDIM   256          // message dim = 2L
#define UDIM   512          // update dim = 4L
#define NSRC   50000
#define NTGT   50000
#define NEDGE  640000
#define SLOPE  0.01f
#define BN_EPS 1e-5f

// ---------------------------------------------------------------------------
// Weight convert + transpose: src row-major [K][N] f32 -> dst [N][K] bf16.
// Transposed storage makes each lane's WMMA B-fragment two contiguous 16B runs of K.
// ---------------------------------------------------------------------------
__global__ void cvt_transpose(const float* __restrict__ src,
                              bf16_t* __restrict__ dst, int K, int N) {
    int i = blockIdx.x * blockDim.x + threadIdx.x;
    if (i >= K * N) return;
    int n = i % N;
    int k = i / N;
    dst[n * K + k] = (bf16_t)src[i];
}

// ---------------------------------------------------------------------------
// Edge MLP: per block = 64 edges, 256 threads (8 waves, 2 Mgroups x 4 Ngroups).
//   A[64x256] = concat(x_s[src], edge_attr)  (bf16 in LDS)
//   T = leaky(A @ W1 + b1);  msg = T @ W2 + b2
//   atomicAdd msg rows into agg[tgt].
// Each wave: 32 rows (2 M-tiles) x 64 cols (4 N-tiles) = 8 accumulators.
// B fragments are loaded once per (k0, nt) and reused across both M-tiles.
// ---------------------------------------------------------------------------
__launch_bounds__(256)
__global__ void edge_mlp(const float* __restrict__ xs,
                         const int*   __restrict__ srcIdx,
                         const int*   __restrict__ tgtIdx,
                         const float* __restrict__ eattr,
                         const bf16_t* __restrict__ W1T,
                         const float* __restrict__ b1,
                         const bf16_t* __restrict__ W2T,
                         const float* __restrict__ b2,
                         float* __restrict__ agg) {
    __shared__ bf16_t Abuf[64][MDIM + 8];   // +8 pad: bank spread, keeps 16B align
    __shared__ bf16_t Tbuf[64][MDIM + 8];
    __shared__ int    sidx[64];
    __shared__ int    tgidx[64];

    const int tid  = threadIdx.x;
    const int lane = tid & 31;
    const int wave = tid >> 5;
    const int e0   = blockIdx.x * 64;

    if (tid < 64) {
        sidx[tid]  = srcIdx[e0 + tid];
        tgidx[tid] = tgtIdx[e0 + tid];
    }
    __syncthreads();

    // Stage A tile (coalesced: 256 consecutive c per 256 threads).
    for (int idx = tid; idx < 64 * MDIM; idx += 256) {
        int r = idx >> 8;          // / 256
        int c = idx & (MDIM - 1);
        float v = (c < LDIM) ? xs[(long)sidx[r] * LDIM + c]
                             : eattr[(long)(e0 + r) * LDIM + (c - LDIM)];
        Abuf[r][c] = (bf16_t)v;
    }
    __syncthreads();

    const int nlo  = lane & 15;
    const int khiA = (lane < 16) ? 0 : 8;    // 16-bit A layout K offset
    const int khiB = (lane < 16) ? 0 : 8;    // 16-bit B layout K offset
    const int mg   = wave >> 2;              // 0..1 : row group (32 rows)
    const int ng   = wave & 3;               // 0..3 : col group (64 cols)
    const int rowBase = mg * 32;
    const int colBase = ng * 64;

    // ---- GEMM1: T = leaky(A @ W1 + b1) ----
    {
        v8f acc[2][4] = {};
        for (int k0 = 0; k0 < MDIM; k0 += 32) {
            v16bf bfrag[4];
#pragma unroll
            for (int nt = 0; nt < 4; ++nt) {
                const bf16_t* bp = &W1T[(long)(colBase + nt * 16 + nlo) * MDIM + k0 + khiB];
                *((v8bf*)&bfrag[nt])       = *((const v8bf*)bp);
                *(((v8bf*)&bfrag[nt]) + 1) = *((const v8bf*)(bp + 16));
            }
#pragma unroll
            for (int mt = 0; mt < 2; ++mt) {
                v16bf a;
                const bf16_t* ap = &Abuf[rowBase + mt * 16 + nlo][k0 + khiA];
                *((v8bf*)&a)       = *((const v8bf*)ap);        // K run 0..7 (+khiA)
                *(((v8bf*)&a) + 1) = *((const v8bf*)(ap + 16)); // K run 16..23 (+khiA)
#pragma unroll
                for (int nt = 0; nt < 4; ++nt) {
                    acc[mt][nt] = __builtin_amdgcn_wmma_f32_16x16x32_bf16(
                        false, a, false, bfrag[nt], (short)0, acc[mt][nt], false, false);
                }
            }
        }
#pragma unroll
        for (int mt = 0; mt < 2; ++mt) {
#pragma unroll
            for (int nt = 0; nt < 4; ++nt) {
                const int col = colBase + nt * 16 + nlo;
                const float bias = b1[col];
#pragma unroll
                for (int j = 0; j < 8; ++j) {
                    float v = acc[mt][nt][j] + bias;
                    v = (v > 0.f) ? v : v * SLOPE;
                    const int row = rowBase + mt * 16 + ((lane < 16) ? j : (j + 8));
                    Tbuf[row][col] = (bf16_t)v;
                }
            }
        }
    }
    __syncthreads();

    // ---- GEMM2: msg = T @ W2 + b2, scatter-add into agg ----
    {
        v8f acc[2][4] = {};
        for (int k0 = 0; k0 < MDIM; k0 += 32) {
            v16bf bfrag[4];
#pragma unroll
            for (int nt = 0; nt < 4; ++nt) {
                const bf16_t* bp = &W2T[(long)(colBase + nt * 16 + nlo) * MDIM + k0 + khiB];
                *((v8bf*)&bfrag[nt])       = *((const v8bf*)bp);
                *(((v8bf*)&bfrag[nt]) + 1) = *((const v8bf*)(bp + 16));
            }
#pragma unroll
            for (int mt = 0; mt < 2; ++mt) {
                v16bf a;
                const bf16_t* ap = &Tbuf[rowBase + mt * 16 + nlo][k0 + khiA];
                *((v8bf*)&a)       = *((const v8bf*)ap);
                *(((v8bf*)&a) + 1) = *((const v8bf*)(ap + 16));
#pragma unroll
                for (int nt = 0; nt < 4; ++nt) {
                    acc[mt][nt] = __builtin_amdgcn_wmma_f32_16x16x32_bf16(
                        false, a, false, bfrag[nt], (short)0, acc[mt][nt], false, false);
                }
            }
        }
#pragma unroll
        for (int mt = 0; mt < 2; ++mt) {
#pragma unroll
            for (int nt = 0; nt < 4; ++nt) {
                const int col = colBase + nt * 16 + nlo;
                const float bias = b2[col];
#pragma unroll
                for (int j = 0; j < 8; ++j) {
                    const int row = rowBase + mt * 16 + ((lane < 16) ? j : (j + 8));
                    const int t   = tgidx[row];
                    atomicAdd(&agg[(long)t * MDIM + col], acc[mt][nt][j] + bias);
                }
            }
        }
    }
}

// ---------------------------------------------------------------------------
// Node update: per block = 16 nodes, 128 threads (4 waves).
//   A[16x512] = concat(x_t, agg, x_u)
//   T = leaky(A @ U1 + c1)  (512 cols: 8 N-tiles per wave)
//   h = T @ U2 + c2         (128 cols: 2 N-tiles per wave) -> h buffer
// ---------------------------------------------------------------------------
__launch_bounds__(128)
__global__ void node_update(const float* __restrict__ xt,
                            const float* __restrict__ agg,
                            const float* __restrict__ xu,
                            const bf16_t* __restrict__ U1T,
                            const float* __restrict__ c1,
                            const bf16_t* __restrict__ U2T,
                            const float* __restrict__ c2,
                            float* __restrict__ hout) {
    __shared__ bf16_t Abuf[16][UDIM + 8];
    __shared__ bf16_t Tbuf[16][UDIM + 8];

    const int tid  = threadIdx.x;
    const int lane = tid & 31;
    const int wave = tid >> 5;
    const int n0   = blockIdx.x * 16;

    for (int idx = tid; idx < 16 * UDIM; idx += 128) {
        int r = idx >> 9;          // / 512
        int c = idx & (UDIM - 1);
        const int node = n0 + r;
        float v;
        if (c < LDIM)             v = xt[(long)node * LDIM + c];
        else if (c < LDIM + MDIM) v = agg[(long)node * MDIM + (c - LDIM)];
        else                      v = xu[c - (LDIM + MDIM)];
        Abuf[r][c] = (bf16_t)v;
    }
    __syncthreads();

    const int nlo  = lane & 15;
    const int khiA = (lane < 16) ? 0 : 8;
    const int khiB = (lane < 16) ? 0 : 8;

    // ---- GEMM1: T = leaky(A @ U1 + c1), 128 cols per wave ----
    {
        v8f acc[8] = {};
        const int colBase = wave * 128;
        for (int k0 = 0; k0 < UDIM; k0 += 32) {
            v16bf a;
            {
                const bf16_t* ap = &Abuf[nlo][k0 + khiA];
                *((v8bf*)&a)       = *((const v8bf*)ap);
                *(((v8bf*)&a) + 1) = *((const v8bf*)(ap + 16));
            }
#pragma unroll
            for (int nt = 0; nt < 8; ++nt) {
                const int col = colBase + nt * 16 + nlo;
                const bf16_t* bp = &U1T[(long)col * UDIM + k0 + khiB];
                v16bf b;
                *((v8bf*)&b)       = *((const v8bf*)bp);
                *(((v8bf*)&b) + 1) = *((const v8bf*)(bp + 16));
                acc[nt] = __builtin_amdgcn_wmma_f32_16x16x32_bf16(
                    false, a, false, b, (short)0, acc[nt], false, false);
            }
        }
#pragma unroll
        for (int nt = 0; nt < 8; ++nt) {
            const int col = colBase + nt * 16 + nlo;
            const float bias = c1[col];
#pragma unroll
            for (int j = 0; j < 8; ++j) {
                float v = acc[nt][j] + bias;
                v = (v > 0.f) ? v : v * SLOPE;
                const int row = (lane < 16) ? j : (j + 8);
                Tbuf[row][col] = (bf16_t)v;
            }
        }
    }
    __syncthreads();

    // ---- GEMM2: h = T @ U2 + c2, 32 cols per wave ----
    {
        v8f acc[2] = {};
        const int colBase = wave * 32;
        for (int k0 = 0; k0 < UDIM; k0 += 32) {
            v16bf a;
            {
                const bf16_t* ap = &Tbuf[nlo][k0 + khiA];
                *((v8bf*)&a)       = *((const v8bf*)ap);
                *(((v8bf*)&a) + 1) = *((const v8bf*)(ap + 16));
            }
#pragma unroll
            for (int nt = 0; nt < 2; ++nt) {
                const int col = colBase + nt * 16 + nlo;
                const bf16_t* bp = &U2T[(long)col * UDIM + k0 + khiB];
                v16bf b;
                *((v8bf*)&b)       = *((const v8bf*)bp);
                *(((v8bf*)&b) + 1) = *((const v8bf*)(bp + 16));
                acc[nt] = __builtin_amdgcn_wmma_f32_16x16x32_bf16(
                    false, a, false, b, (short)0, acc[nt], false, false);
            }
        }
#pragma unroll
        for (int nt = 0; nt < 2; ++nt) {
            const int col = colBase + nt * 16 + nlo;
            const float bias = c2[col];
#pragma unroll
            for (int j = 0; j < 8; ++j) {
                const int row = (lane < 16) ? j : (j + 8);
                hout[(long)(n0 + row) * LDIM + col] = acc[nt][j] + bias;
            }
        }
    }
}

// ---------------------------------------------------------------------------
// BatchNorm: column-wise sum / sum-of-squares over NT rows (128 threads =
// one column each, coalesced row-major reads), then normalize.
// ---------------------------------------------------------------------------
__global__ void bn_stats(const float* __restrict__ h,
                         float* __restrict__ sums,
                         float* __restrict__ sumsq,
                         int rowsPerBlock) {
    const int col = threadIdx.x;                 // 128 threads
    const int r0  = blockIdx.x * rowsPerBlock;
    int r1 = r0 + rowsPerBlock;
    if (r1 > NTGT) r1 = NTGT;
    float s = 0.f, sq = 0.f;
    for (int r = r0; r < r1; ++r) {
        const float v = h[(long)r * LDIM + col];
        s  += v;
        sq += v * v;
    }
    atomicAdd(&sums[col],  s);
    atomicAdd(&sumsq[col], sq);
}

__global__ void bn_apply(const float* __restrict__ h,
                         const float* __restrict__ sums,
                         const float* __restrict__ sumsq,
                         const float* __restrict__ gamma,
                         const float* __restrict__ beta,
                         float* __restrict__ out) {
    const long i = (long)blockIdx.x * blockDim.x + threadIdx.x;
    if (i >= (long)NTGT * LDIM) return;
    const int c = (int)(i & (LDIM - 1));
    const float inv_n = 1.0f / (float)NTGT;
    const float mu  = sums[c] * inv_n;
    const float var = sumsq[c] * inv_n - mu * mu;
    out[i] = gamma[c] * (h[i] - mu) * rsqrtf(var + BN_EPS) + beta[c];
}

// ---------------------------------------------------------------------------
// Launch
// ---------------------------------------------------------------------------
extern "C" void kernel_launch(void* const* d_in, const int* in_sizes, int n_in,
                              void* d_out, int out_size, void* d_ws, size_t ws_size,
                              hipStream_t stream) {
    const float* x_s   = (const float*)d_in[0];
    const float* x_t   = (const float*)d_in[1];
    const int*   eidx  = (const int*)  d_in[2];   // [2, E] flat
    const float* eattr = (const float*)d_in[3];
    const float* x_u   = (const float*)d_in[4];
    const float* W1    = (const float*)d_in[5];
    const float* b1    = (const float*)d_in[6];
    const float* W2    = (const float*)d_in[7];
    const float* b2    = (const float*)d_in[8];
    const float* U1    = (const float*)d_in[9];
    const float* c1    = (const float*)d_in[10];
    const float* U2    = (const float*)d_in[11];
    const float* c2    = (const float*)d_in[12];
    const float* gamma = (const float*)d_in[13];
    const float* beta  = (const float*)d_in[14];
    float* out = (float*)d_out;

    // Workspace layout (256B aligned)
    char* ws = (char*)d_ws;
    size_t off = 0;
    float*  agg  = (float*)(ws + off);  off += (size_t)NTGT * MDIM * sizeof(float);   // 51.2 MB
    float*  hbuf = (float*)(ws + off);  off += (size_t)NTGT * LDIM * sizeof(float);   // 25.6 MB
    bf16_t* W1T  = (bf16_t*)(ws + off); off += (size_t)MDIM * MDIM * sizeof(bf16_t);
    bf16_t* W2T  = (bf16_t*)(ws + off); off += (size_t)MDIM * MDIM * sizeof(bf16_t);
    bf16_t* U1T  = (bf16_t*)(ws + off); off += (size_t)UDIM * UDIM * sizeof(bf16_t);
    bf16_t* U2T  = (bf16_t*)(ws + off); off += (size_t)UDIM * LDIM * sizeof(bf16_t);
    float*  sums  = (float*)(ws + off); off += LDIM * sizeof(float);
    float*  sumsq = (float*)(ws + off); off += LDIM * sizeof(float);
    (void)ws_size; (void)in_sizes; (void)n_in; (void)out_size;

    // Zero accumulation buffers (capture-legal async memsets)
    hipMemsetAsync(agg,  0, (size_t)NTGT * MDIM * sizeof(float), stream);
    hipMemsetAsync(sums, 0, 2 * LDIM * sizeof(float), stream);

    // Weight conversion (f32 -> bf16, transposed [N][K])
    cvt_transpose<<<(MDIM * MDIM + 255) / 256, 256, 0, stream>>>(W1, W1T, MDIM, MDIM);
    cvt_transpose<<<(MDIM * MDIM + 255) / 256, 256, 0, stream>>>(W2, W2T, MDIM, MDIM);
    cvt_transpose<<<(UDIM * UDIM + 255) / 256, 256, 0, stream>>>(U1, U1T, UDIM, UDIM);
    cvt_transpose<<<(UDIM * LDIM + 255) / 256, 256, 0, stream>>>(U2, U2T, UDIM, LDIM);

    // Edge MLP + scatter-add (E/64 = 10000 blocks)
    edge_mlp<<<NEDGE / 64, 256, 0, stream>>>(x_s, eidx, eidx + NEDGE, eattr,
                                             W1T, b1, W2T, b2, agg);

    // Node update (NT/16 = 3125 blocks)
    node_update<<<NTGT / 16, 128, 0, stream>>>(x_t, agg, x_u, U1T, c1, U2T, c2, hbuf);

    // BatchNorm (400 blocks x 125 rows = 50000)
    bn_stats<<<400, 128, 0, stream>>>(hbuf, sums, sumsq, 125);
    bn_apply<<<((long)NTGT * LDIM + 255) / 256, 256, 0, stream>>>(hbuf, sums, sumsq,
                                                                  gamma, beta, out);
}
